// FrameAugment_39771397161402
// MI455X (gfx1250) — matrix operand order repaired and checked by reading, not compiled, not run on hardware
//
#include <hip/hip_runtime.h>

// Problem constants (from the reference): B=16, L=2048, F=512, all fp32.
#define BB 16
#define LL 2048
#define FF 512

typedef __attribute__((ext_vector_type(16))) _Float16 v16h;
typedef __attribute__((ext_vector_type(8)))  float    v8f;

// ---------------------------------------------------------------------------
// Kernel 1: per-row softmax statistics over noise (B*L rows of length L).
// Writes float2 {row_max, 1/sum(exp(x-row_max))} per row (256 KB total).
// ---------------------------------------------------------------------------
__global__ void __launch_bounds__(256) fa_rowstats(const float* __restrict__ noise,
                                                   float2* __restrict__ stats) {
    const int row = blockIdx.x;                       // 0 .. B*L-1
    const float* p = noise + (size_t)row * LL;
    const int t = threadIdx.x;                        // 256 threads
    const int lane = t & 31, wave = t >> 5;

    float x[8];
#pragma unroll
    for (int i = 0; i < 8; ++i) x[i] = p[t + i * 256];

    float m = x[0];
#pragma unroll
    for (int i = 1; i < 8; ++i) m = fmaxf(m, x[i]);
#pragma unroll
    for (int off = 16; off > 0; off >>= 1) m = fmaxf(m, __shfl_xor(m, off, 32));

    __shared__ float redm[8];
    if (lane == 0) redm[wave] = m;
    __syncthreads();
    float bm = redm[0];
#pragma unroll
    for (int w = 1; w < 8; ++w) bm = fmaxf(bm, redm[w]);

    float s = 0.f;
#pragma unroll
    for (int i = 0; i < 8; ++i) s += __expf(x[i] - bm);
#pragma unroll
    for (int off = 16; off > 0; off >>= 1) s += __shfl_xor(s, off, 32);

    __shared__ float reds[8];
    if (lane == 0) reds[wave] = s;
    __syncthreads();
    if (t == 0) {
        float bs = 0.f;
#pragma unroll
        for (int w = 0; w < 8; ++w) bs += reds[w];
        stats[row] = make_float2(bm, 1.0f / bs);
    }
}

// ---------------------------------------------------------------------------
// Kernel 2: feature (B,L,F) f32  ->  Xt (B,F,L) f16 via LDS tile transpose.
// Makes the WMMA B-fragment a contiguous 32-byte load per lane.
// ---------------------------------------------------------------------------
__global__ void __launch_bounds__(256) fa_transpose_f16(const float* __restrict__ feat,
                                                        _Float16* __restrict__ xt) {
    __shared__ float tile[32][33];                    // +1 pad: no bank conflicts
    const int b  = blockIdx.z;
    const int f0 = blockIdx.x * 32;
    const int j0 = blockIdx.y * 32;
    const int tx = threadIdx.x;                       // 0..31
    const int ty = threadIdx.y;                       // 0..7

#pragma unroll
    for (int r = ty; r < 32; r += 8)
        tile[r][tx] = feat[((size_t)b * LL + (j0 + r)) * FF + f0 + tx];
    __syncthreads();
#pragma unroll
    for (int r = ty; r < 32; r += 8)
        xt[((size_t)b * FF + (f0 + r)) * LL + j0 + tx] = (_Float16)tile[tx][r];
}

// ---------------------------------------------------------------------------
// Kernel 3: fused softmax-apply + GEMM via v_wmma_f32_16x16x32_f16.
//   out[b,i,f] = sum_j exp(noise[b,i,j]-max_i)*inv_sum_i * feature[b,j,f]
// 8 waves (2 along M x 4 along N); wave tile 64x64 = 4x4 WMMA fragments.
// Workgroup tile 128 (M) x 256 (N); K stepped by 32.
// ---------------------------------------------------------------------------
__global__ void __launch_bounds__(256, 1)
fa_gemm_wmma(const float* __restrict__ noise, const _Float16* __restrict__ xt,
             const float2* __restrict__ stats, float* __restrict__ out) {
    const int b     = blockIdx.z;
    const int lane  = threadIdx.x & 31;
    const int wave  = threadIdx.x >> 5;
    const int lrow  = lane & 15;                      // M/N index within 16
    const int lhalf = lane >> 4;                      // which half-wave
    const int mw    = wave & 1;                       // 0..1
    const int nw    = wave >> 1;                      // 0..3
    const int mBase = blockIdx.x * 128 + mw * 64;
    const int nBase = blockIdx.y * 256 + nw * 64;

    // Per-lane row pointers + softmax stats for the 4 M-subtiles.
    const float* np[4];
    float rmax[4], rinv[4];
#pragma unroll
    for (int m = 0; m < 4; ++m) {
        const int row = mBase + m * 16 + lrow;        // A layout: M = lane&15
        const float2 st = stats[(size_t)b * LL + row];
        rmax[m] = st.x;
        rinv[m] = st.y;
        np[m]   = noise + ((size_t)b * LL + row) * LL;
    }
    // Per-lane column pointers into transposed f16 feature (B-fragment source).
    const _Float16* xp[4];
#pragma unroll
    for (int n = 0; n < 4; ++n) {
        const int col = nBase + n * 16 + lrow;        // B layout: N = lane&15
        xp[n] = xt + ((size_t)b * FF + col) * LL + lhalf * 16; // K half per half-wave
    }

    v8f acc[4][4];
    const v8f vzero = {};
#pragma unroll
    for (int m = 0; m < 4; ++m)
#pragma unroll
        for (int n = 0; n < 4; ++n) acc[m][n] = vzero;

    for (int k0 = 0; k0 < LL; k0 += 32) {
        // B fragments: 32x16 f16; lane holds 16 consecutive K of its column.
        v16h bf[4];
#pragma unroll
        for (int n = 0; n < 4; ++n)
            bf[n] = *(const v16h*)(xp[n] + k0);

#pragma unroll
        for (int m = 0; m < 4; ++m) {
            // A fragment 16x32 f16. Lanes 0-15: K k0+0..15,
            // lanes 16-31: K k0+16..31 (contiguous, matches B fragment).
            const float* rp = np[m] + k0 + lhalf * 16;
            const float4 x0 = *(const float4*)(rp);
            const float4 x1 = *(const float4*)(rp + 4);
            const float4 x2 = *(const float4*)(rp + 8);
            const float4 x3 = *(const float4*)(rp + 12);
            const float xs[16] = {x0.x, x0.y, x0.z, x0.w, x1.x, x1.y, x1.z, x1.w,
                                  x2.x, x2.y, x2.z, x2.w, x3.x, x3.y, x3.z, x3.w};
            const float mx = rmax[m], iv = rinv[m];
            v16h a;
#pragma unroll
            for (int i = 0; i < 16; ++i)
                a[i] = (_Float16)(__expf(xs[i] - mx) * iv);

#pragma unroll
            for (int n = 0; n < 4; ++n)
                acc[m][n] = __builtin_amdgcn_wmma_f32_16x16x32_f16(
                    /*neg_a=*/false, a, /*neg_b=*/false, bf[n],
                    /*c_mod=*/(short)0, acc[m][n],
                    /*reuse_a=*/false, /*reuse_b=*/false);
        }

        if (k0 + 32 < LL) {
#pragma unroll
            for (int m = 0; m < 4; ++m)
                __builtin_prefetch(np[m] + k0 + 32 + lhalf * 8, 0, 1);
        }
    }

    // Store: f32 C/D layout -> VGPR r holds (row r | r+8) x (N = lane&15).
#pragma unroll
    for (int m = 0; m < 4; ++m)
#pragma unroll
        for (int n = 0; n < 4; ++n) {
            const int col = nBase + n * 16 + lrow;
#pragma unroll
            for (int r = 0; r < 8; ++r) {
                const int row = mBase + m * 16 + lhalf * 8 + r;
                out[((size_t)b * LL + row) * FF + col] = acc[m][n][r];
            }
        }
}

// ---------------------------------------------------------------------------
// Host launcher. d_in[0] = feature (B,L,F) f32, d_in[1] = noise (B,L,L) f32.
// ws usage: [0, 256KB)  float2 row stats
//           [256KB, 256KB+32MiB) f16 transposed feature (B,F,L)
// ---------------------------------------------------------------------------
extern "C" void kernel_launch(void* const* d_in, const int* in_sizes, int n_in,
                              void* d_out, int out_size, void* d_ws, size_t ws_size,
                              hipStream_t stream) {
    const float* feature = (const float*)d_in[0];
    const float* noise   = (const float*)d_in[1];
    float* out           = (float*)d_out;

    float2*    stats = (float2*)d_ws;
    _Float16*  xt    = (_Float16*)((char*)d_ws + (size_t)BB * LL * sizeof(float2));

    // 1. Row softmax stats: one block per row.
    fa_rowstats<<<dim3(BB * LL), dim3(256), 0, stream>>>(noise, stats);

    // 2. Transpose + f16 convert feature: (B,L,F) -> (B,F,L).
    fa_transpose_f16<<<dim3(FF / 32, LL / 32, BB), dim3(32, 8), 0, stream>>>(feature, xt);

    // 3. Fused softmax-apply GEMM with WMMA.
    fa_gemm_wmma<<<dim3(LL / 128, FF / 256, BB), dim3(256), 0, stream>>>(noise, xt, stats, out);
}
